// SNCGNetwork_74663711474064
// MI455X (gfx1250) — compile-verified, matrix operand order checked by reading, not run on hardware
//
#include <hip/hip_runtime.h>

typedef _Float16 v8h  __attribute__((ext_vector_type(8)));
typedef _Float16 v16h __attribute__((ext_vector_type(16)));
typedef float    v8f  __attribute__((ext_vector_type(8)));
typedef int      v4i  __attribute__((ext_vector_type(4)));

typedef __attribute__((address_space(1))) v4i* gv4i_p;
typedef __attribute__((address_space(3))) v4i* lv4i_p;

#define B_SZ    1024
#define D_IN    1024
#define D_H     4096
#define D_OUT   1024
#define T_STEPS 32

#define DT_TAU_MEM 0.1f   // DT * TAU_MEM_INV = 1e-3 * 100
#define DT_TAU_SYN 0.2f   // DT * TAU_SYN_INV = 1e-3 * 200
#define V_TH 1.0f

// ---------------- CDNA5 async-copy helpers ----------------

__device__ __forceinline__ void async_copy_b128(const _Float16* g, _Float16* l) {
#if __has_builtin(__builtin_amdgcn_global_load_async_to_lds_b128)
  __attribute__((address_space(1))) void* g1 =
      (__attribute__((address_space(1))) void*)g;
  __attribute__((address_space(3))) void* l3 =
      (__attribute__((address_space(3))) void*)l;
  __builtin_amdgcn_global_load_async_to_lds_b128((gv4i_p)g1, (lv4i_p)l3, 0, 0);
#else
  unsigned lds_off = (unsigned)(unsigned long long)
      (__attribute__((address_space(3))) void*)l;
  asm volatile("global_load_async_to_lds_b128 %0, %1, off"
               :: "v"(lds_off), "v"((unsigned long long)g)
               : "memory");
#endif
}

__device__ __forceinline__ void wait_async0() {
#if __has_builtin(__builtin_amdgcn_s_wait_asynccnt)
  __builtin_amdgcn_s_wait_asynccnt(0);
#else
  asm volatile("s_wait_asynccnt 0x0" ::: "memory");
#endif
}

// ---------------- elementwise kernels ----------------

__global__ void f32_to_f16_kernel(const float* __restrict__ src,
                                  _Float16* __restrict__ dst, int n) {
  int i = blockIdx.x * blockDim.x + threadIdx.x;
  if (i < n) dst[i] = (_Float16)src[i];
}

// one step of the constant-current encoder (sequential in t, parallel in B*D_IN)
__global__ void encode_step_kernel(const float* __restrict__ x,
                                   float* __restrict__ venc,
                                   _Float16* __restrict__ spk) {
  int i = blockIdx.x * blockDim.x + threadIdx.x;
  float v = venc[i];
  v = v + DT_TAU_MEM * ((0.0f - v) + x[i]);
  float z = (v > V_TH) ? 1.0f : 0.0f;
  venc[i] = v - z * v;
  spk[i] = (_Float16)z;
}

// ---------------- fused GEMM (X @ W^T) + LIF step ----------------
// A: [M,K] f16 row-major (spikes), W: [N,K] f16 row-major.
// 8 waves arranged 4(M) x 2(N); per-wave tile (TM*16) x (TN*16).
// Double-buffered LDS staged via GLOBAL_LOAD_ASYNC_TO_LDS_B128.
template<int BM, int BN, int BK, bool OUT_LAYER>
__global__ __launch_bounds__(256)
void lif_gemm_kernel(const _Float16* __restrict__ A,
                     const _Float16* __restrict__ W,
                     float* __restrict__ Vst, float* __restrict__ Ist,
                     _Float16* __restrict__ Zout,
                     float* __restrict__ Cnt,
                     float* __restrict__ Out,
                     int M, int N, int K, int is_last)
{
  constexpr int LDT = BK + 8;            // +16B row pad -> conflict-free ds_load_b128
  constexpr int TM  = BM / 64;           // subtiles per wave, M (4 waves in M)
  constexpr int TN  = BN / 32;           // subtiles per wave, N (2 waves in N)
  __shared__ _Float16 As[2][BM * LDT];
  __shared__ _Float16 Bs[2][BN * LDT];

  const int tid  = threadIdx.x;
  const int lane = tid & 31;
  const int wave = tid >> 5;
  const int wm   = wave >> 1;            // 0..3
  const int wn   = wave & 1;             // 0..1

  const int blockM = blockIdx.y * BM;
  const int blockN = blockIdx.x * BN;

  const int l16   = lane & 15;
  const int lhalf = lane >> 4;           // 0|1
  const int kfb   = lhalf << 3;          // per-lane K base: 0 | 8 (ISA A/B frag layout)

  v8f acc[TM][TN] = {};

  auto issue_stage = [&](int buf, int kk) {
    #pragma unroll
    for (int it = 0; it < (BM * BK / 8) / 256; ++it) {
      int c = tid + it * 256;
      int row = c / (BK / 8), kc = c % (BK / 8);
      async_copy_b128(&A[(size_t)(blockM + row) * K + kk + kc * 8],
                      &As[buf][row * LDT + kc * 8]);
    }
    #pragma unroll
    for (int it = 0; it < (BN * BK / 8) / 256; ++it) {
      int c = tid + it * 256;
      int row = c / (BK / 8), kc = c % (BK / 8);
      async_copy_b128(&W[(size_t)(blockN + row) * K + kk + kc * 8],
                      &Bs[buf][row * LDT + kc * 8]);
    }
  };

  const int nstages = K / BK;
  issue_stage(0, 0);

  for (int s = 0; s < nstages; ++s) {
    const int cur = s & 1;
    wait_async0();        // my async loads for stage s are in LDS
    __syncthreads();      // everyone's are; prior stage's readers are done

    if (s + 1 < nstages) issue_stage(1 - cur, (s + 1) * BK);  // overlap with compute

    const _Float16* Ab = As[cur];
    const _Float16* Bb = Bs[cur];

    #pragma unroll
    for (int k32 = 0; k32 < BK; k32 += 32) {
      v16h afrag[TM], bfrag[TN];
      #pragma unroll
      for (int tm = 0; tm < TM; ++tm) {
        int row = wm * (TM * 16) + tm * 16 + l16;
        v8h lo = *(const v8h*)&Ab[row * LDT + k32 + kfb];
        v8h hi = *(const v8h*)&Ab[row * LDT + k32 + kfb + 16];
        afrag[tm] = __builtin_shufflevector(lo, hi, 0,1,2,3,4,5,6,7,8,9,10,11,12,13,14,15);
      }
      #pragma unroll
      for (int tn = 0; tn < TN; ++tn) {
        int row = wn * (TN * 16) + tn * 16 + l16;
        v8h lo = *(const v8h*)&Bb[row * LDT + k32 + kfb];
        v8h hi = *(const v8h*)&Bb[row * LDT + k32 + kfb + 16];
        bfrag[tn] = __builtin_shufflevector(lo, hi, 0,1,2,3,4,5,6,7,8,9,10,11,12,13,14,15);
      }
      #pragma unroll
      for (int tm = 0; tm < TM; ++tm)
        #pragma unroll
        for (int tn = 0; tn < TN; ++tn)
          acc[tm][tn] = __builtin_amdgcn_wmma_f32_16x16x32_f16(
              false, afrag[tm], false, bfrag[tn],
              (short)0, acc[tm][tn], false, false);
    }
    __syncthreads();      // done reading As[cur]/Bs[cur] before it is overwritten
  }

  // ---- fused LIF epilogue (D layout: VGPR r, lane -> M = 8*(lane>>4)+r, N = lane&15) ----
  #pragma unroll
  for (int tm = 0; tm < TM; ++tm) {
    #pragma unroll
    for (int tn = 0; tn < TN; ++tn) {
      int colg = blockN + wn * (TN * 16) + tn * 16 + l16;
      int row0 = blockM + wm * (TM * 16) + tm * 16 + lhalf * 8;
      size_t base = (size_t)row0 * N + colg;
      float vv[8], ci[8];
      #pragma unroll
      for (int r = 0; r < 8; ++r) vv[r] = Vst[base + (size_t)r * N];
      #pragma unroll
      for (int r = 0; r < 8; ++r) ci[r] = Ist[base + (size_t)r * N];
      float cc[8];
      if (OUT_LAYER) {
        #pragma unroll
        for (int r = 0; r < 8; ++r) cc[r] = Cnt[base + (size_t)r * N];
      }
      #pragma unroll
      for (int r = 0; r < 8; ++r) {
        size_t idx = base + (size_t)r * N;
        float a     = acc[tm][tn][r];
        float v_dec = vv[r] + DT_TAU_MEM * ((0.0f - vv[r]) + ci[r]);
        float i_new = ci[r] - DT_TAU_SYN * ci[r] + a;
        float z     = (v_dec > V_TH) ? 1.0f : 0.0f;
        Vst[idx] = (1.0f - z) * v_dec;
        Ist[idx] = i_new;
        if (OUT_LAYER) {
          float c = cc[r] + z;
          Cnt[idx] = c;
          if (is_last) Out[idx] = c * (1.0f / (float)T_STEPS);
        } else {
          Zout[idx] = (_Float16)z;
        }
      }
    }
  }
}

// ---------------- host launch ----------------

extern "C" void kernel_launch(void* const* d_in, const int* in_sizes, int n_in,
                              void* d_out, int out_size, void* d_ws, size_t ws_size,
                              hipStream_t stream) {
  const float* x  = (const float*)d_in[0];   // [B, D_IN]
  const float* W1 = (const float*)d_in[1];   // [D_H, D_IN]
  const float* W2 = (const float*)d_in[2];   // [D_OUT, D_H]
  float* out = (float*)d_out;                // [B, D_OUT]

  char* ws = (char*)d_ws;
  size_t off = 0;
  auto carve = [&](size_t bytes) -> void* {
    void* p = ws + off;
    off += (bytes + 255) & ~(size_t)255;
    return p;
  };

  _Float16* W1h  = (_Float16*)carve((size_t)D_H * D_IN * 2);
  _Float16* W2h  = (_Float16*)carve((size_t)D_OUT * D_H * 2);
  _Float16* spk  = (_Float16*)carve((size_t)B_SZ * D_IN * 2);
  _Float16* z1   = (_Float16*)carve((size_t)B_SZ * D_H * 2);
  float*    venc = (float*)carve((size_t)B_SZ * D_IN * 4);
  float*    v1   = (float*)carve((size_t)B_SZ * D_H * 4);
  float*    i1   = (float*)carve((size_t)B_SZ * D_H * 4);
  float*    v2   = (float*)carve((size_t)B_SZ * D_OUT * 4);
  float*    i2   = (float*)carve((size_t)B_SZ * D_OUT * 4);
  float*    cnt  = (float*)carve((size_t)B_SZ * D_OUT * 4);

  // zero all sequential state (deterministic per call)
  (void)hipMemsetAsync(venc, 0, (size_t)B_SZ * D_IN * 4, stream);
  (void)hipMemsetAsync(v1,   0, (size_t)B_SZ * D_H * 4, stream);
  (void)hipMemsetAsync(i1,   0, (size_t)B_SZ * D_H * 4, stream);
  (void)hipMemsetAsync(v2,   0, (size_t)B_SZ * D_OUT * 4, stream);
  (void)hipMemsetAsync(i2,   0, (size_t)B_SZ * D_OUT * 4, stream);
  (void)hipMemsetAsync(cnt,  0, (size_t)B_SZ * D_OUT * 4, stream);

  // convert weights to f16 once per call
  {
    int n1 = D_H * D_IN, n2 = D_OUT * D_H;
    f32_to_f16_kernel<<<(n1 + 255) / 256, 256, 0, stream>>>(W1, W1h, n1);
    f32_to_f16_kernel<<<(n2 + 255) / 256, 256, 0, stream>>>(W2, W2h, n2);
  }

  // layer1: 128x128 block tile (TM=2,TN=4) -> grid 32x8 = 256 blocks
  // layer2: 128x64  block tile (TM=2,TN=2) -> grid 16x8 = 128 blocks
  dim3 grid1(D_H  / 128, B_SZ / 128);
  dim3 grid2(D_OUT / 64, B_SZ / 128);

  for (int t = 0; t < T_STEPS; ++t) {
    encode_step_kernel<<<(B_SZ * D_IN) / 256, 256, 0, stream>>>(x, venc, spk);

    lif_gemm_kernel<128, 128, 64, false><<<grid1, 256, 0, stream>>>(
        spk, W1h, v1, i1, z1, nullptr, nullptr, B_SZ, D_H, D_IN, 0);

    lif_gemm_kernel<128, 64, 64, true><<<grid2, 256, 0, stream>>>(
        z1, W2h, v2, i2, nullptr, cnt, out, B_SZ, D_OUT, D_H,
        (t == T_STEPS - 1) ? 1 : 0);
  }
}
